// NCALoss_50818053046733
// MI455X (gfx1250) — compile-verified, matrix operand order checked by reading, not run on hardware
//
#include <hip/hip_runtime.h>
#include <hip/hip_bf16.h>
#include <math.h>

#define NMAX 8192
#define NCA_ALPHA 16.0f
#define BLOCK_THREADS 256
#define WAVES_PER_BLOCK (BLOCK_THREADS / 32)
#define ROWS_PER_BLOCK WAVES_PER_BLOCK

typedef __attribute__((ext_vector_type(2)))  float    v2f;
typedef __attribute__((ext_vector_type(2)))  int      v2i;
typedef __attribute__((ext_vector_type(8)))  float    v8f;
typedef __attribute__((ext_vector_type(16))) _Float16 v16h;

__device__ __forceinline__ float waveReduceSumF(float v) {
#pragma unroll
    for (int m = 16; m >= 1; m >>= 1) v += __shfl_xor(v, m, 32);
    return v;
}
__device__ __forceinline__ float waveReduceMaxF(float v) {
#pragma unroll
    for (int m = 16; m >= 1; m >>= 1) v = fmaxf(v, __shfl_xor(v, m, 32));
    return v;
}
__device__ __forceinline__ int waveReduceSumI(int v) {
#pragma unroll
    for (int m = 16; m >= 1; m >>= 1) v += __shfl_xor(v, m, 32);
    return v;
}

// ---------------------------------------------------------------------------
// Kernel 1: one wave per row. Stage x + labels into LDS with CDNA5 async
// global->LDS loads (ASYNCcnt), then two LDS-resident passes over the row
// using 64-bit vectorized LDS reads (2 pairs per lane per iteration).
// ---------------------------------------------------------------------------
__global__ __launch_bounds__(BLOCK_THREADS) void nca_rows_kernel(
    const float* __restrict__ x, const int* __restrict__ t,
    float* __restrict__ lossRow, int n) {
    __shared__ __align__(16) float sx[NMAX];
    __shared__ __align__(16) int   st[NMAX];

    const int tid = threadIdx.x;

    // ---- async stage x[0..n) and t[0..n) into LDS (b128 = 4 elems/lane) ----
    {
        unsigned sxBase = (unsigned)(uintptr_t)&sx[0];
        unsigned stBase = (unsigned)(uintptr_t)&st[0];
        const int nv4 = n >> 2;
        for (int c = tid; c < nv4; c += BLOCK_THREADS) {
            unsigned lx = sxBase + (unsigned)c * 16u;
            unsigned long long gx = (unsigned long long)(uintptr_t)(x + c * 4);
            asm volatile("global_load_async_to_lds_b128 %0, %1, off"
                         :: "v"(lx), "v"(gx) : "memory");
            unsigned lt = stBase + (unsigned)c * 16u;
            unsigned long long gt = (unsigned long long)(uintptr_t)(t + c * 4);
            asm volatile("global_load_async_to_lds_b128 %0, %1, off"
                         :: "v"(lt), "v"(gt) : "memory");
        }
        // tail (n not multiple of 4)
        for (int e = (nv4 << 2) + tid; e < n; e += BLOCK_THREADS) {
            unsigned lx = sxBase + (unsigned)e * 4u;
            unsigned long long gx = (unsigned long long)(uintptr_t)(x + e);
            asm volatile("global_load_async_to_lds_b32 %0, %1, off"
                         :: "v"(lx), "v"(gx) : "memory");
            unsigned lt = stBase + (unsigned)e * 4u;
            unsigned long long gt = (unsigned long long)(uintptr_t)(t + e);
            asm volatile("global_load_async_to_lds_b32 %0, %1, off"
                         :: "v"(lt), "v"(gt) : "memory");
        }
        asm volatile("s_wait_asynccnt 0" ::: "memory");
    }
    __syncthreads();

    const int lane = tid & 31;
    const int wid  = tid >> 5;
    const int row  = blockIdx.x * ROWS_PER_BLOCK + wid;
    if (row >= n) return;

    const float xi = sx[row];
    const int   ti = st[row];

    const v2f* sx2 = (const v2f*)sx;
    const v2i* st2 = (const v2i*)st;
    const int  nv2 = n >> 1;
    const bool odd = (n & 1) != 0;

    // ---- pass 1: thresh = max over (pos|neg) pairs; base = mean of row ----
    float mx  = -INFINITY;
    float sum = 0.0f;
    {
        auto acc1 = [&](float xj, int tj) {
            float s = fabsf(xi - xj);
            bool same = (tj == ti);
            sum += s;
            float cand = (!same || (s < 1.0f)) ? s : -INFINITY;
            mx = fmaxf(mx, cand);
        };
#pragma unroll 4
        for (int p = lane; p < nv2; p += 32) {
            v2f xv = sx2[p];
            v2i tv = st2[p];
            acc1(xv[0], tv[0]);
            acc1(xv[1], tv[1]);
        }
        if (odd && lane == 0) acc1(sx[n - 1], st[n - 1]);
    }
    const float base   = waveReduceSumF(sum) / (float)n;
    const float thresh = waveReduceMaxF(mx);

    // ---- pass 2: exp-weighted masked sums ----
    // w = exp(alpha*(base-s)) = exp2( fma(-alpha*log2e, s, alpha*base*log2e) )
    const float LOG2E = 1.4426950408889634f;
    const float c1 =  NCA_ALPHA * base * LOG2E;
    const float c2 = -NCA_ALPHA * LOG2E;

    float pa = 0.0f;  // sum w over full pos_mask (fallback)
    float pn = 0.0f;  // sum w over pos_mask & below
    float ng = 0.0f;  // sum w over neg_mask & below
    int   pcnt = 0;   // count of pos_mask & below (has_pos)
    {
        auto acc2 = [&](float xj, int tj) {
            float s = fabsf(xi - xj);
            bool same = (tj == ti);
            float w = __builtin_amdgcn_exp2f(fmaf(c2, s, c1));
            bool posm  = same && (s < 1.0f);
            bool below = (s < thresh);
            bool pnsel = posm && below;
            pa += posm  ? w : 0.0f;
            pn += pnsel ? w : 0.0f;
            pcnt += pnsel ? 1 : 0;
            ng += (!same && below) ? w : 0.0f;
        };
#pragma unroll 4
        for (int p = lane; p < nv2; p += 32) {
            v2f xv = sx2[p];
            v2i tv = st2[p];
            acc2(xv[0], tv[0]);
            acc2(xv[1], tv[1]);
        }
        if (odd && lane == 0) acc2(sx[n - 1], st[n - 1]);
    }
    pa   = waveReduceSumF(pa);
    pn   = waveReduceSumF(pn);
    ng   = waveReduceSumF(ng);
    pcnt = waveReduceSumI(pcnt);

    if (lane == 0) {
        float pos = (pcnt > 0) ? pn : pa;
        lossRow[row] = -logf(pos / (pos + ng));
    }
}

// ---------------------------------------------------------------------------
// Kernel 2: single wave. WMMA-based exact reduction of the per-row losses
// (sum + precision count), then last-row pos/neg mean sims.
// With A == all-ones, sum(all entries of D) == 16 * sum(all entries of B),
// independent of the WMMA VGPR striping -> exact & layout-agnostic.
// ---------------------------------------------------------------------------
__global__ __launch_bounds__(32) void nca_finalize_kernel(
    const float* __restrict__ x, const int* __restrict__ t,
    const float* __restrict__ lossRow, float* __restrict__ out, int n) {
    const int lane = threadIdx.x;

    float sL = 0.0f, sP = 0.0f;   // scaled by 16 until the end
    int it = 0;
#if __has_builtin(__builtin_amdgcn_wmma_f32_16x16x4_f32)
    {
        v8f accL = {0.f,0.f,0.f,0.f,0.f,0.f,0.f,0.f};
        v8f accP = {0.f,0.f,0.f,0.f,0.f,0.f,0.f,0.f};
        v2f ones; ones[0] = 1.0f; ones[1] = 1.0f;
        for (; it + 64 <= n; it += 64) {
            int j = it + lane * 2;
            v2f bl; bl[0] = lossRow[j]; bl[1] = lossRow[j + 1];
            v2f bp; bp[0] = (bl[0] < 0.6f) ? 1.0f : 0.0f;
                    bp[1] = (bl[1] < 0.6f) ? 1.0f : 0.0f;
            accL = __builtin_amdgcn_wmma_f32_16x16x4_f32(
                false, ones, false, bl, (short)0, accL, false, false);
            accP = __builtin_amdgcn_wmma_f32_16x16x4_f32(
                false, ones, false, bp, (short)0, accP, false, false);
        }
#pragma unroll
        for (int c = 0; c < 8; ++c) { sL += accL[c]; sP += accP[c]; }
    }
#else
    {
        v8f accL = {0.f,0.f,0.f,0.f,0.f,0.f,0.f,0.f};
        v8f accP = {0.f,0.f,0.f,0.f,0.f,0.f,0.f,0.f};
        v16h ones;
#pragma unroll
        for (int c = 0; c < 16; ++c) ones[c] = (_Float16)1.0f;
        for (; it + 512 <= n; it += 512) {
            int j = it + lane * 16;
            v16h bl, bp;
#pragma unroll
            for (int c = 0; c < 16; ++c) {
                float lv = lossRow[j + c];
                bl[c] = (_Float16)lv;
                bp[c] = (_Float16)((lv < 0.6f) ? 1.0f : 0.0f);
            }
            accL = __builtin_amdgcn_wmma_f32_16x16x32_f16(
                false, ones, false, bl, (short)0, accL, false, false);
            accP = __builtin_amdgcn_wmma_f32_16x16x32_f16(
                false, ones, false, bp, (short)0, accP, false, false);
        }
#pragma unroll
        for (int c = 0; c < 8; ++c) { sL += accL[c]; sP += accP[c]; }
    }
#endif
    // scalar tail (n not multiple of the WMMA chunk), kept at the x16 scale
    for (int j = it + lane; j < n; j += 32) {
        float lv = lossRow[j];
        sL += 16.0f * lv;
        sP += 16.0f * ((lv < 0.6f) ? 1.0f : 0.0f);
    }
    sL = waveReduceSumF(sL) * (1.0f / 16.0f);
    sP = waveReduceSumF(sP) * (1.0f / 16.0f);

    // ---- last-row (n-1) pos/neg mean similarity ----
    const float xi = x[n - 1];
    const int   ti = t[n - 1];
    float ps = 0.0f, ns = 0.0f;
    int pc = 0, nc = 0;
    for (int j = lane; j < n; j += 32) {
        float s = fabsf(xi - x[j]);
        if (t[j] == ti) {
            if (s < 1.0f) { ps += s; pc++; }
        } else {
            ns += s; nc++;
        }
    }
    ps = waveReduceSumF(ps);
    ns = waveReduceSumF(ns);
    pc = waveReduceSumI(pc);
    nc = waveReduceSumI(nc);

    if (lane == 0) {
        out[0] = sL / (float)n;          // loss
        out[1] = sP / (float)n;          // prec
        out[2] = ps / (float)pc;         // mean_pos_sim (row n-1)
        out[3] = ns / (float)nc;         // mean_neg_sim (row n-1)
    }
}

extern "C" void kernel_launch(void* const* d_in, const int* in_sizes, int n_in,
                              void* d_out, int out_size, void* d_ws, size_t ws_size,
                              hipStream_t stream) {
    const float* x = (const float*)d_in[0];
    const int*   t = (const int*)d_in[1];
    float* out = (float*)d_out;
    int n = in_sizes[0];
    if (n > NMAX) n = NMAX;

    float* lossRow = (float*)d_ws;   // n floats of scratch

    int blocks = (n + ROWS_PER_BLOCK - 1) / ROWS_PER_BLOCK;
    nca_rows_kernel<<<blocks, BLOCK_THREADS, 0, stream>>>(x, t, lossRow, n);
    nca_finalize_kernel<<<1, 32, 0, stream>>>(x, t, lossRow, out, n);
}
